// GAT_57629871178587
// MI455X (gfx1250) — matrix operand order repaired and checked by reading, not compiled, not run on hardware
//
#include <hip/hip_runtime.h>
#include <hip/hip_bf16.h>

// ---------------------------------------------------------------------------
// 3-layer single-head GATConv for MI455X (gfx1250, wave32).
//  - Projections z = h @ W use v_wmma_f32_16x16x32_bf16 (bf16 in, f32 acc).
//  - Segment softmax over dst via ordered-uint atomicMax + f32 atomicAdd.
//  - Scatter out[dst] += alpha * z[src] with 64 threads per edge (L2-resident).
// ---------------------------------------------------------------------------

typedef __attribute__((ext_vector_type(16))) __bf16 v16bf;
typedef __attribute__((ext_vector_type(8)))  float  v8f;

union GatFrag { v16bf v; uint4 q[2]; };

__device__ __forceinline__ float gat_lrelu(float x, float s) {
    return x > 0.0f ? x : x * s;
}

// Order-preserving encode of f32 into u32 so atomicMax(u32) == float max.
__device__ __forceinline__ unsigned gat_enc(float f) {
    unsigned u = __float_as_uint(f);
    return (u & 0x80000000u) ? ~u : (u | 0x80000000u);
}
__device__ __forceinline__ float gat_dec(unsigned u) {
    return (u & 0x80000000u) ? __uint_as_float(u & 0x7FFFFFFFu)
                             : __uint_as_float(~u);
}

// --- conversions ------------------------------------------------------------

__global__ void gat_cvt_x(const float* __restrict__ x,
                          __hip_bfloat16* __restrict__ hb, int n) {
    int i = blockIdx.x * blockDim.x + threadIdx.x;
    if (i < n) hb[i] = __float2bfloat16(x[i]);
}

// W: [K][64] row-major f32  ->  WbT: [64][K] bf16 (transposed for B-frag loads)
__global__ void gat_cvt_wT(const float* __restrict__ W,
                           __hip_bfloat16* __restrict__ WbT, int K) {
    int i = blockIdx.x * blockDim.x + threadIdx.x;
    if (i >= K * 64) return;
    int r = i >> 6, c = i & 63;
    WbT[c * K + r] = __float2bfloat16(W[i]);
}

// --- per-layer init ---------------------------------------------------------

__global__ void gat_init(float* __restrict__ outAcc, float* __restrict__ denom,
                         unsigned* __restrict__ mEnc, int Nn) {
    int i = blockIdx.x * blockDim.x + threadIdx.x;
    if (i < Nn * 64) outAcc[i] = 0.0f;
    if (i < Nn) { denom[i] = 0.0f; mEnc[i] = 0u; }
}

// --- z = hb @ W  (bf16 WMMA, f32 accumulate) --------------------------------
// grid.x = Nn/16 blocks, 128 threads (4 waves); wave w owns output cols
// [16w,16w+16). A per ISA 16-bit A layout; B read from WbT rows (contiguous K).

__global__ void __launch_bounds__(128)
gat_gemm_wmma(const __hip_bfloat16* __restrict__ A,
              const __hip_bfloat16* __restrict__ BT,
              float* __restrict__ Z, int K, int Nn) {
    int m0 = blockIdx.x * 16;
    if (m0 >= Nn) return;                  // uniform per block: EXEC stays ~0
    int wave = threadIdx.x >> 5;
    int lane = threadIdx.x & 31;
    int hf   = lane >> 4;                  // half-wave select
    int mn   = lane & 15;                  // A-row / B-col owned by this lane
    int n0   = wave * 16;

    const __hip_bfloat16* arow = A  + (size_t)(m0 + mn) * K;
    const __hip_bfloat16* brow = BT + (size_t)(n0 + mn) * K;

    v8f acc = {};
    for (int k0 = 0; k0 < K; k0 += 32) {
        GatFrag a, b;
        // A 16x32 bf16: lanes0-15 K:{k0..k0+7, k0+16..k0+23}, lanes16-31 +8
        a.q[0] = *(const uint4*)(arow + k0 + hf * 8);
        a.q[1] = *(const uint4*)(arow + k0 + 16 + hf * 8);
        // B 32x16 bf16 (col per lane): lanes0-15 K:{k0..k0+7, k0+16..k0+23}, lanes16-31 +8
        b.q[0] = *(const uint4*)(brow + k0 + hf * 8);
        b.q[1] = *(const uint4*)(brow + k0 + 16 + hf * 8);
        acc = __builtin_amdgcn_wmma_f32_16x16x32_bf16(
            false, a.v, false, b.v, (short)0, acc, false, false);
    }
    // C/D: VGPR r -> row (r + 8*hf), col mn
#pragma unroll
    for (int r = 0; r < 8; ++r)
        Z[(size_t)(m0 + r + 8 * hf) * 64 + n0 + mn] = acc[r];
}

// --- per-node attention scores: e_src[i]=z[i]·a_src, e_dst[i]=z[i]·a_dst ----

__global__ void gat_escore(const float* __restrict__ z,
                           const float* __restrict__ aS,
                           const float* __restrict__ aD,
                           float* __restrict__ esrc, float* __restrict__ edst,
                           int Nn) {
    int wave = (blockIdx.x * blockDim.x + threadIdx.x) >> 5;
    int lane = threadIdx.x & 31;
    if (wave >= Nn) return;
    float z0 = z[(size_t)wave * 64 + lane];
    float z1 = z[(size_t)wave * 64 + lane + 32];
    float ps = z0 * aS[lane] + z1 * aS[lane + 32];
    float pd = z0 * aD[lane] + z1 * aD[lane + 32];
#pragma unroll
    for (int off = 16; off > 0; off >>= 1) {
        ps += __shfl_xor(ps, off, 32);
        pd += __shfl_xor(pd, off, 32);
    }
    if (lane == 0) { esrc[wave] = ps; edst[wave] = pd; }
}

// --- edge kernels -----------------------------------------------------------

__device__ __forceinline__ void gat_edge_sd(const int* ei, int E0, int idx,
                                            int& s, int& d) {
    if (idx < E0) { s = ei[idx]; d = ei[E0 + idx]; }
    else          { s = idx - E0; d = s; }           // self-loops appended
}

__global__ void gat_edge_max(const int* __restrict__ ei, int E0, int ET,
                             const float* __restrict__ esrc,
                             const float* __restrict__ edst,
                             float* __restrict__ e, unsigned* __restrict__ mEnc) {
    int idx = blockIdx.x * blockDim.x + threadIdx.x;
    if (idx >= ET) return;
    int s, d; gat_edge_sd(ei, E0, idx, s, d);
    float v = gat_lrelu(esrc[s] + edst[d], 0.2f);
    e[idx] = v;
    atomicMax(&mEnc[d], gat_enc(v));
}

__global__ void gat_edge_exp(const int* __restrict__ ei, int E0, int ET,
                             float* __restrict__ e,
                             const unsigned* __restrict__ mEnc,
                             float* __restrict__ denom) {
    int idx = blockIdx.x * blockDim.x + threadIdx.x;
    if (idx >= ET) return;
    int s, d; gat_edge_sd(ei, E0, idx, s, d);
    float ex = expf(e[idx] - gat_dec(mEnc[d]));
    e[idx] = ex;
    atomicAdd(&denom[d], ex);
}

// 64 threads per edge: out[d,f] += alpha * z[s,f]
__global__ void gat_edge_scatter(const int* __restrict__ ei, int E0, int ET,
                                 const float* __restrict__ ex,
                                 const float* __restrict__ denom,
                                 const float* __restrict__ z,
                                 float* __restrict__ outAcc,
                                 float* __restrict__ outAlpha) {
    int t = blockIdx.x * blockDim.x + threadIdx.x;
    int edge = t >> 6, f = t & 63;
    if (edge >= ET) return;
    int s, d; gat_edge_sd(ei, E0, edge, s, d);
    float alpha = ex[edge] / (denom[d] + 1e-16f);
    if (outAlpha != nullptr && f == 0) outAlpha[edge] = alpha;
    atomicAdd(&outAcc[(size_t)d * 64 + f], alpha * z[(size_t)s * 64 + f]);
}

// --- bias + leaky_relu(0.01); feeds next layer as bf16 or final output ------

__global__ void gat_bias_act(const float* __restrict__ outAcc,
                             const float* __restrict__ bias,
                             float* __restrict__ outH,
                             __hip_bfloat16* __restrict__ hbNext,
                             int last, int Nn) {
    int i = blockIdx.x * blockDim.x + threadIdx.x;
    if (i >= Nn * 64) return;
    float v = gat_lrelu(outAcc[i] + bias[i & 63], 0.01f);
    if (last) outH[i] = v;
    else      hbNext[i] = __float2bfloat16(v);
}

// --- ei output (reference returns [2, E+N] with self-loops appended) --------

__global__ void gat_write_ei(const int* __restrict__ ei, int E0, int ET,
                             float* __restrict__ outEi) {
    int idx = blockIdx.x * blockDim.x + threadIdx.x;
    if (idx >= ET) return;
    int s, d; gat_edge_sd(ei, E0, idx, s, d);
    outEi[idx]      = (float)s;
    outEi[ET + idx] = (float)d;
}

// ---------------------------------------------------------------------------

static inline int cdiv(long long a, int b) { return (int)((a + b - 1) / b); }

static void gat_run_layer(hipStream_t stream,
                          const __hip_bfloat16* hb, int K,
                          const float* W, const float* aS, const float* aD,
                          const float* bias,
                          const int* ei, int E0, int ET, int Nn,
                          __hip_bfloat16* WbT, float* z, float* esrc,
                          float* edst, unsigned* mEnc, float* denom,
                          float* ebuf, float* outAcc,
                          float* outH, __hip_bfloat16* hbNext,
                          float* outAlpha, int last) {
    gat_cvt_wT<<<cdiv(K * 64, 256), 256, 0, stream>>>(W, WbT, K);
    gat_init<<<cdiv((long long)Nn * 64, 256), 256, 0, stream>>>(outAcc, denom, mEnc, Nn);
    gat_gemm_wmma<<<cdiv(Nn, 16), 128, 0, stream>>>(hb, WbT, z, K, Nn);
    gat_escore<<<cdiv((long long)Nn * 32, 256), 256, 0, stream>>>(z, aS, aD, esrc, edst, Nn);
    gat_edge_max<<<cdiv(ET, 256), 256, 0, stream>>>(ei, E0, ET, esrc, edst, ebuf, mEnc);
    gat_edge_exp<<<cdiv(ET, 256), 256, 0, stream>>>(ei, E0, ET, ebuf, mEnc, denom);
    gat_edge_scatter<<<cdiv((long long)ET * 64, 256), 256, 0, stream>>>(
        ei, E0, ET, ebuf, denom, z, outAcc, outAlpha);
    gat_bias_act<<<cdiv((long long)Nn * 64, 256), 256, 0, stream>>>(
        outAcc, bias, outH, hbNext, last, Nn);
}

extern "C" void kernel_launch(void* const* d_in, const int* in_sizes, int n_in,
                              void* d_out, int out_size, void* d_ws, size_t ws_size,
                              hipStream_t stream) {
    const float* x    = (const float*)d_in[0];
    const int*   ei   = (const int*)  d_in[1];
    const float* W1   = (const float*)d_in[2];
    const float* aS1  = (const float*)d_in[3];
    const float* aD1  = (const float*)d_in[4];
    const float* b1   = (const float*)d_in[5];
    const float* W2   = (const float*)d_in[6];
    const float* aS2  = (const float*)d_in[7];
    const float* aD2  = (const float*)d_in[8];
    const float* b2   = (const float*)d_in[9];
    const float* W3   = (const float*)d_in[10];
    const float* aS3  = (const float*)d_in[11];
    const float* aD3  = (const float*)d_in[12];
    const float* b3   = (const float*)d_in[13];

    const int Nn = in_sizes[0] / 128;   // 50000
    const int E0 = in_sizes[1] / 2;     // 800000
    const int ET = E0 + Nn;             // edges incl. self-loops

    // ---- workspace carve (all 256B aligned) ----
    char* p = (char*)d_ws;
    auto take = [&](size_t bytes) {
        char* q = p;
        p += (bytes + 255) & ~(size_t)255;
        return q;
    };
    __hip_bfloat16* hb    = (__hip_bfloat16*)take((size_t)Nn * 128 * 2);
    __hip_bfloat16* WbT   = (__hip_bfloat16*)take((size_t)128 * 64 * 2);
    float*          z     = (float*)take((size_t)Nn * 64 * 4);
    float*          esrc  = (float*)take((size_t)Nn * 4);
    float*          edst  = (float*)take((size_t)Nn * 4);
    unsigned*       mEnc  = (unsigned*)take((size_t)Nn * 4);
    float*          denom = (float*)take((size_t)Nn * 4);
    float*          ebuf  = (float*)take((size_t)ET * 4);
    float*          outAc = (float*)take((size_t)Nn * 64 * 4);

    // ---- d_out layout: h [N*64] | ei [2*ET] | alpha [ET], all as f32 ----
    float* outH     = (float*)d_out;
    float* outEi    = outH + (size_t)Nn * 64;
    float* outAlpha = outEi + 2 * (size_t)ET;

    gat_write_ei<<<cdiv(ET, 256), 256, 0, stream>>>(ei, E0, ET, outEi);
    gat_cvt_x<<<cdiv((long long)Nn * 128, 256), 256, 0, stream>>>(x, hb, Nn * 128);

    // layer 1: K=128 ; hb (N x 128) -> hb (N x 64, overwritten after gemm)
    gat_run_layer(stream, hb, 128, W1, aS1, aD1, b1, ei, E0, ET, Nn,
                  WbT, z, esrc, edst, mEnc, denom, ebuf, outAc,
                  nullptr, hb, nullptr, 0);
    // layer 2: K=64
    gat_run_layer(stream, hb, 64, W2, aS2, aD2, b2, ei, E0, ET, Nn,
                  WbT, z, esrc, edst, mEnc, denom, ebuf, outAc,
                  nullptr, hb, nullptr, 0);
    // layer 3: K=64 ; final -> d_out h + alpha
    gat_run_layer(stream, hb, 64, W3, aS3, aD3, b3, ei, E0, ET, Nn,
                  WbT, z, esrc, edst, mEnc, denom, ebuf, outAc,
                  outH, nullptr, outAlpha, 1);
}
